// ReadoutSimple_24988119728559
// MI455X (gfx1250) — matrix operand order repaired and checked by reading, not compiled, not run on hardware
//
#include <hip/hip_runtime.h>
#include <math.h>

// ---------------- problem constants (match reference) ----------------
constexpr int N_ATOM  = 200000;
constexpr int N_FRAG  = 500000;
constexpr int E_EDGES = 500000;   // per edge type
constexpr int IN_F    = 64;
constexpr int HID     = 128;
constexpr int N_HID   = 3;

// ---------------- workspace layout (bytes) ----------------
// acc  : N_FRAG * 64 * 4   = 128,000,000   (fused mean-sum hf, f32)
// cnt  : 2 * N_FRAG * 4    =   4,000,000   (per-etype in-degree, f32)
// w0t  : 128 * 64  * 2     =      16,384   (W_in  transposed, f16, [N][K])
// wht  : 3*128*128 * 2     =      98,304   (Wh[i] transposed, f16, [N][K])
constexpr size_t OFS_CNT = 128000000;
constexpr size_t OFS_W0T = 132000000;
constexpr size_t OFS_WHT = 132016384;

typedef __attribute__((ext_vector_type(16))) _Float16 v16h;
typedef __attribute__((ext_vector_type(8)))  _Float16 v8h;
typedef __attribute__((ext_vector_type(4)))  _Float16 v4h;
typedef __attribute__((ext_vector_type(8)))  float    v8f;

// ---------------- kernel 0: zero acc + cnt (contiguous 132MB) ----------------
__global__ void k_zero(float4* p, int n4) {
  int i = blockIdx.x * blockDim.x + threadIdx.x;
  if (i < n4) p[i] = make_float4(0.f, 0.f, 0.f, 0.f);
}

// ---------------- kernel 1: convert + transpose weights to f16 N-major ------
__global__ void k_wconv(const float* __restrict__ W_in, const float* __restrict__ Wh,
                        _Float16* __restrict__ w0t, _Float16* __restrict__ wht) {
  int t = blockIdx.x * 256 + threadIdx.x;
  if (t < HID * IN_F) {                     // w0t[n][k] = W_in[k][n]
    int n = t >> 6, k = t & 63;
    w0t[t] = (_Float16)W_in[k * HID + n];
  }
  int u = t - HID * IN_F;
  if (u >= 0 && u < N_HID * HID * HID) {    // wht[i][n][k] = Wh[i][k][n]
    int i = u >> 14, rem = u & 16383;
    int n = rem >> 7, k = rem & 127;
    wht[u] = (_Float16)Wh[(i << 14) + (k << 7) + n];
  }
}

// ---------------- kernel 2: per-(etype,fragment) edge counts ----------------
__global__ void k_count(const int* __restrict__ dst, float* __restrict__ cnt) {
  int t = blockIdx.x * 256 + threadIdx.x;
  if (t >= 2 * E_EDGES) return;
  int etype = (t >= E_EDGES) ? 1 : 0;
  __hip_atomic_fetch_add(&cnt[etype * N_FRAG + dst[t]], 1.0f,
                         __ATOMIC_RELAXED, __HIP_MEMORY_SCOPE_AGENT);
}

// ---------------- kernel 3: scaled scatter-add (fused mean + etype sum) -----
// 16 lanes per edge, 4 feats per lane; acc[d][:] += h[s][:] / cnt[etype][d]
__global__ void k_scatter(const float* __restrict__ h, const int* __restrict__ src,
                          const int* __restrict__ dst, const float* __restrict__ cnt,
                          float* __restrict__ acc) {
  int t = blockIdx.x * 256 + threadIdx.x;   // exactly 2*E*16 = 16,000,000
  int edge = t >> 4;
  int part = t & 15;
  int etype = (edge >= E_EDGES) ? 1 : 0;
  int s = src[edge];
  int d = dst[edge];
  float invc = 1.0f / cnt[etype * N_FRAG + d];   // >= 1 for any present edge
  const float4 v = *(const float4*)(h + (size_t)s * 64 + part * 4);
  float* a = acc + (size_t)d * 64 + part * 4;
  __hip_atomic_fetch_add(a + 0, v.x * invc, __ATOMIC_RELAXED, __HIP_MEMORY_SCOPE_AGENT);
  __hip_atomic_fetch_add(a + 1, v.y * invc, __ATOMIC_RELAXED, __HIP_MEMORY_SCOPE_AGENT);
  __hip_atomic_fetch_add(a + 2, v.z * invc, __ATOMIC_RELAXED, __HIP_MEMORY_SCOPE_AGENT);
  __hip_atomic_fetch_add(a + 3, v.w * invc, __ATOMIC_RELAXED, __HIP_MEMORY_SCOPE_AGENT);
}

// ---------------- kernel 4: fused MLP via v_wmma_f32_16x16x32_f16 -----------
// One workgroup (8 waves) per 16-fragment tile. Wave w owns output columns
// [16w, 16w+16). Activations ping-pong through LDS in f16; accumulate in f32.
__global__ __launch_bounds__(256)
void k_mlp(const float* __restrict__ acc,
           const _Float16* __restrict__ w0t, const _Float16* __restrict__ wht,
           const float* __restrict__ b_in, const float* __restrict__ bh,
           const float* __restrict__ wout, const float* __restrict__ b_out,
           float* __restrict__ out) {
  constexpr int KP = 136;                    // row pitch in halves (16B-aligned)
  __shared__ _Float16 act[2][16][KP];
  __shared__ float red[256];

  const int tid  = threadIdx.x;
  const int lane = tid & 31;
  const int wv   = tid >> 5;                 // 0..7
  const int n0   = wv * 16;
  const int m0   = blockIdx.x * 16;          // N_FRAG % 16 == 0
  const int col  = lane & 15;                // C/D: N = col ; A: row M = col
  const int hi   = lane >> 4;                // lane half selects K sub-block

  // stage hf tile [16 x 64] f32 -> f16 into act[0] (coalesced float4 per lane)
  {
    float4 v = *(const float4*)(acc + (size_t)m0 * 64 + tid * 4);
    int r = tid >> 4, k = (tid & 15) * 4;
    v4h hv; hv.x = (_Float16)v.x; hv.y = (_Float16)v.y;
            hv.z = (_Float16)v.z; hv.w = (_Float16)v.w;
    *(v4h*)&act[0][r][k] = hv;
  }
  __syncthreads();

  int cur = 0;

  // ---- layer 0: 64 -> 128, tanh ----
  {
    v8f c;
    float b = b_in[n0 + col];
    #pragma unroll
    for (int j = 0; j < 8; ++j) c[j] = b;
    #pragma unroll
    for (int kb = 0; kb < IN_F / 32; ++kb) {
      union { v16h v; v8h h[2]; } A;
      int abase = kb * 32 + hi * 16;         // A layout: K 0-15 | 16-31
      A.h[0] = *(const v8h*)&act[cur][col][abase];
      A.h[1] = *(const v8h*)&act[cur][col][abase + 8];
      v16h B = *(const v16h*)(w0t + (n0 + col) * IN_F + kb * 32 + hi * 16);
      c = __builtin_amdgcn_wmma_f32_16x16x32_f16(false, A.v, false, B,
                                                 (short)0, c, false, false);
    }
    #pragma unroll
    for (int j = 0; j < 8; ++j) {            // C layout: VGPR j -> M = j + 8*hi
      int row = j + hi * 8;
      act[cur ^ 1][row][n0 + col] = (_Float16)tanhf(c[j]);
    }
  }
  __syncthreads();
  cur ^= 1;

  // ---- hidden layers: 128 -> 128, ReLU ----
  for (int l = 0; l < N_HID; ++l) {
    const _Float16* lw = wht + l * HID * HID;
    v8f c;
    float b = bh[l * HID + n0 + col];
    #pragma unroll
    for (int j = 0; j < 8; ++j) c[j] = b;
    #pragma unroll
    for (int kb = 0; kb < HID / 32; ++kb) {
      union { v16h v; v8h h[2]; } A;
      int abase = kb * 32 + hi * 16;
      A.h[0] = *(const v8h*)&act[cur][col][abase];
      A.h[1] = *(const v8h*)&act[cur][col][abase + 8];
      v16h B = *(const v16h*)(lw + (n0 + col) * HID + kb * 32 + hi * 16);
      c = __builtin_amdgcn_wmma_f32_16x16x32_f16(false, A.v, false, B,
                                                 (short)0, c, false, false);
    }
    #pragma unroll
    for (int j = 0; j < 8; ++j) {
      int row = j + hi * 8;
      float v = c[j];
      act[cur ^ 1][row][n0 + col] = (_Float16)(v > 0.f ? v : 0.f);
    }
    __syncthreads();
    cur ^= 1;
  }

  // ---- output layer: 128 -> 1 (VALU dot + LDS reduction) ----
  {
    int r = tid >> 4, part = tid & 15;
    float p = 0.f;
    #pragma unroll
    for (int j = 0; j < 8; ++j) {
      int k = part * 8 + j;
      p += (float)act[cur][r][k] * wout[k];
    }
    red[tid] = p;
    __syncthreads();
    if (tid < 16) {
      float s = 0.f;
      #pragma unroll
      for (int i = 0; i < 16; ++i) s += red[tid * 16 + i];
      out[m0 + tid] = s + b_out[0];
    }
  }
}

// ---------------- launch ----------------
extern "C" void kernel_launch(void* const* d_in, const int* in_sizes, int n_in,
                              void* d_out, int out_size, void* d_ws, size_t ws_size,
                              hipStream_t stream) {
  const float* h     = (const float*)d_in[0];
  const float* W_in  = (const float*)d_in[1];
  const float* b_in  = (const float*)d_in[2];
  const float* Wh    = (const float*)d_in[3];
  const float* bh    = (const float*)d_in[4];
  const float* W_out = (const float*)d_in[5];
  const float* b_out = (const float*)d_in[6];
  const int*   src   = (const int*)d_in[7];
  const int*   dst   = (const int*)d_in[8];
  float* out = (float*)d_out;

  char* ws = (char*)d_ws;
  float*    acc = (float*)ws;
  float*    cnt = (float*)(ws + OFS_CNT);
  _Float16* w0t = (_Float16*)(ws + OFS_W0T);
  _Float16* wht = (_Float16*)(ws + OFS_WHT);

  // zero acc + cnt (33,000,000 floats contiguous -> 8,250,000 float4)
  {
    int n4 = (N_FRAG * 64 + 2 * N_FRAG) / 4;
    k_zero<<<(n4 + 255) / 256, 256, 0, stream>>>((float4*)ws, n4);
  }
  // f16 transposed weights (8192 + 49152 elements)
  k_wconv<<<(HID * IN_F + N_HID * HID * HID + 255) / 256, 256, 0, stream>>>(
      W_in, Wh, w0t, wht);
  // per-etype in-degree
  k_count<<<(2 * E_EDGES + 255) / 256, 256, 0, stream>>>(dst, cnt);
  // fused mean scatter (2*E*16 threads, exact multiple of 256)
  k_scatter<<<(2 * E_EDGES * 16) / 256, 256, 0, stream>>>(h, src, dst, cnt, acc);
  // WMMA MLP: one 16-fragment tile per workgroup
  k_mlp<<<N_FRAG / 16, 256, 0, stream>>>(acc, w0t, wht, b_in, bh, W_out, b_out, out);
}
